// EdgeFeatures_81484119539777
// MI455X (gfx1250) — matrix operand order repaired and checked by reading, not compiled, not run on hardware
//
#include <hip/hip_runtime.h>

#define C_IN   128
#define C_HID  256
#define EPS_IN 1e-5f

typedef _Float16 v16h __attribute__((ext_vector_type(16)));
typedef _Float16 v8h  __attribute__((ext_vector_type(8)));
typedef float    v8f  __attribute__((ext_vector_type(8)));

// ---------------- workspace layout ----------------
// halves: [0,32768)      W1a^T  (n-major [256][128])
//         [32768,65536)  W1b^T  (n-major [256][128])
//         [65536,131072) W2cat^T (n-major [128][512]; k<256 from W2a, k>=256 from W2b)
// bytes 262144: floats b1a[256], b1b[256], b2sum[128]
#define WS_BIAS_BYTES 262144

// ---------------- LDS layout ----------------
#define LDS_W1A_H   0
#define LDS_W1B_H   34816      // 256*136
#define LDS_W2_H    69632      // + 256*136 ; W2cat occupies 128*520 halves
#define LDS_BIAS_B  272384     // (69632 + 66560) * 2 bytes
#define LDS_SCR_B   274944     // + 640 floats
#define LDS_TOTAL_B 295424     // + 16 waves * 16*40 halves * 2B

#define THREADS 512            // 16 wave32s -> 4 waves/SIMD resident
#define EDGES_PER_BLOCK 256

__device__ __forceinline__ v8f wmma16(v16h a, v16h b, v8f c) {
  return __builtin_amdgcn_wmma_f32_16x16x32_f16(false, a, false, b, (short)0, c,
                                                false, false);
}

// 16 contiguous halves (two 16B LDS loads)
__device__ __forceinline__ v16h ld16c(const _Float16* p) {
  v8h a = *(const v8h*)(p);
  v8h b = *(const v8h*)(p + 8);
  return __builtin_shufflevector(a, b, 0,1,2,3,4,5,6,7,8,9,10,11,12,13,14,15);
}
// halves p[0..7] and p[16..23] (A-fragment K split)
__device__ __forceinline__ v16h ld16s(const _Float16* p) {
  v8h a = *(const v8h*)(p);
  v8h b = *(const v8h*)(p + 16);
  return __builtin_shufflevector(a, b, 0,1,2,3,4,5,6,7,8,9,10,11,12,13,14,15);
}

// ---------------- weight prep: fp32 -> f16, transpose, fuse ----------------
__global__ void gnn_prep_kernel(const float* __restrict__ W1a, const float* __restrict__ b1a,
                                const float* __restrict__ W2a, const float* __restrict__ b2a,
                                const float* __restrict__ W1b, const float* __restrict__ b1b,
                                const float* __restrict__ W2b, const float* __restrict__ b2b,
                                _Float16* __restrict__ wsH, float* __restrict__ wsB) {
  int tid = blockIdx.x * blockDim.x + threadIdx.x;
  int stride = gridDim.x * blockDim.x;
  for (int idx = tid; idx < 32768; idx += stride) {         // [256 n][128 k]
    int n = idx >> 7, k = idx & 127;
    wsH[idx]         = (_Float16)W1a[k * C_HID + n];
    wsH[32768 + idx] = (_Float16)W1b[k * C_HID + n];
  }
  for (int idx = tid; idx < 65536; idx += stride) {         // [128 n][512 k]
    int n = idx >> 9, k = idx & 511;
    float v = (k < 256) ? W2a[k * C_IN + n] : W2b[(k - 256) * C_IN + n];
    wsH[65536 + idx] = (_Float16)v;
  }
  for (int idx = tid; idx < 256; idx += stride) {
    wsB[idx]       = b1a[idx];
    wsB[256 + idx] = b1b[idx];
  }
  for (int idx = tid; idx < 128; idx += stride)
    wsB[512 + idx] = b2a[idx] + b2b[idx];
}

// One 32-wide hidden chunk: GEMM1 (2 tiles, K=128) -> bias+ReLU -> re-layout
// through per-wave LDS scratch -> fused GEMM2 K-chunk into 8 accumulators.
__device__ __forceinline__ void mlp_chunk(
    v16h A0, v16h A1, v16h A2, v16h A3,
    const _Float16* w1,      // LDS W1 base (n-major, stride 136)
    const float* bb,         // LDS bias (256 floats)
    const _Float16* w2,      // LDS W2cat base (n-major, stride 520)
    _Float16* scr,           // per-wave scratch: 16 rows x 40 halves
    int cc, int c, int col, int hi, v8f (&accO)[8]) {

  const _Float16* b0p = w1 + (size_t)(cc * 32 + col)      * 136 + hi * 16;
  const _Float16* b1p = w1 + (size_t)(cc * 32 + 16 + col) * 136 + hi * 16;

  v8f h0, h1;
#pragma unroll
  for (int r = 0; r < 8; ++r) { h0[r] = 0.0f; h1[r] = 0.0f; }

  v16h B00 = ld16c(b0p);        v16h B01 = ld16c(b1p);
  v16h B10 = ld16c(b0p + 32);   v16h B11 = ld16c(b1p + 32);
  h0 = wmma16(A0, B00, h0);     h1 = wmma16(A0, B01, h1);
  v16h B20 = ld16c(b0p + 64);   v16h B21 = ld16c(b1p + 64);
  h0 = wmma16(A1, B10, h0);     h1 = wmma16(A1, B11, h1);
  v16h B30 = ld16c(b0p + 96);   v16h B31 = ld16c(b1p + 96);
  h0 = wmma16(A2, B20, h0);     h1 = wmma16(A2, B21, h1);
  h0 = wmma16(A3, B30, h0);     h1 = wmma16(A3, B31, h1);

  // bias + ReLU, park as f16 in scratch (D-layout -> A-layout re-shape)
  float bv0 = bb[cc * 32 + col];
  float bv1 = bb[cc * 32 + 16 + col];
#pragma unroll
  for (int r = 0; r < 8; ++r) {
    int M = r + hi * 8;
    scr[M * 40 + col]      = (_Float16)fmaxf(h0[r] + bv0, 0.0f);
    scr[M * 40 + 16 + col] = (_Float16)fmaxf(h1[r] + bv1, 0.0f);
  }
  // same-wave LDS ops are in-order: read back as A fragment
  v16h Ah = ld16c(scr + col * 40 + hi * 16);

  // fused second GEMM: K-chunk c of the 512-wide concat hidden, 8 N tiles
  const _Float16* w2p = w2 + (size_t)col * 520 + c * 32 + hi * 16;
  v16h C0 = ld16c(w2p);
  v16h C1 = ld16c(w2p + 1 * 8320);
  v16h C2 = ld16c(w2p + 2 * 8320);
  v16h C3 = ld16c(w2p + 3 * 8320);
  accO[0] = wmma16(Ah, C0, accO[0]);
  accO[1] = wmma16(Ah, C1, accO[1]);
  v16h C4 = ld16c(w2p + 4 * 8320);
  v16h C5 = ld16c(w2p + 5 * 8320);
  accO[2] = wmma16(Ah, C2, accO[2]);
  accO[3] = wmma16(Ah, C3, accO[3]);
  v16h C6 = ld16c(w2p + 6 * 8320);
  v16h C7 = ld16c(w2p + 7 * 8320);
  accO[4] = wmma16(Ah, C4, accO[4]);
  accO[5] = wmma16(Ah, C5, accO[5]);
  accO[6] = wmma16(Ah, C6, accO[6]);
  accO[7] = wmma16(Ah, C7, accO[7]);
}

// ---------------- fused edge-MLP + InstanceNorm + residual ----------------
__global__ __launch_bounds__(THREADS, 1)
void gnn_edge_kernel(const float* __restrict__ nf, const float* __restrict__ ef,
                     const long long* __restrict__ ei,
                     const _Float16* __restrict__ wsH, const float* __restrict__ wsB,
                     float* __restrict__ out, int nE) {
  extern __shared__ char smem[];
  _Float16* ldsH = (_Float16*)smem;
  float*    ldsB = (float*)(smem + LDS_BIAS_B);
  _Float16* scrA = (_Float16*)(smem + LDS_SCR_B);

  __builtin_prefetch(wsH + (threadIdx.x << 6), 0, 0);   // global_prefetch_b8

  // ---- stage weights + biases into LDS, 16B vector chunks ----
  // W1a / W1b: 32768 halves each = 4096 x v8h ; rows of 128 halves (16 chunks)
  for (int idx = threadIdx.x; idx < 4096; idx += THREADS) {
    int n = idx >> 4, kc = (idx & 15) * 8;
    *(v8h*)(ldsH + LDS_W1A_H + n * 136 + kc) = *(const v8h*)(wsH + idx * 8);
    *(v8h*)(ldsH + LDS_W1B_H + n * 136 + kc) = *(const v8h*)(wsH + 32768 + idx * 8);
  }
  // W2cat: 65536 halves = 8192 x v8h ; rows of 512 halves (64 chunks)
  for (int idx = threadIdx.x; idx < 8192; idx += THREADS) {
    int n = idx >> 6, kc = (idx & 63) * 8;
    *(v8h*)(ldsH + LDS_W2_H + n * 520 + kc) = *(const v8h*)(wsH + 65536 + idx * 8);
  }
  // biases: 640 floats = 160 x float4
  for (int idx = threadIdx.x; idx < 160; idx += THREADS)
    *(float4*)(ldsB + idx * 4) = *(const float4*)(wsB + idx * 4);
  __syncthreads();

  const int wave = threadIdx.x >> 5;
  const int lane = threadIdx.x & 31;
  const int col  = lane & 15;
  const int hi   = lane >> 4;

  const int eBase = blockIdx.x * EDGES_PER_BLOCK + wave * 16;

  // ---- build A fragments for s = nf[src]+nf[dst] and for ef (f16) ----
  int e  = eBase + col;
  int ec = (e < nE) ? e : (nE - 1);
  long long si = ei[ec];
  long long di = ei[(long long)nE + ec];
  const float* ps = nf + (size_t)si * C_IN;
  const float* pd = nf + (size_t)di * C_IN;
  const float* pe = ef + (size_t)ec * C_IN;

  v16h As[4], Ae[4];
#pragma unroll
  for (int kt = 0; kt < 4; ++kt) {
    int k0 = kt * 32 + hi * 16;
    v16h a, b;
#pragma unroll
    for (int j = 0; j < 8; ++j) {
      a[j]     = (_Float16)(ps[k0 + j]      + pd[k0 + j]);
      a[j + 8] = (_Float16)(ps[k0 + 8 + j] + pd[k0 + 8 + j]);
      b[j]     = (_Float16)pe[k0 + j];
      b[j + 8] = (_Float16)pe[k0 + 8 + j];
    }
    As[kt] = a;
    Ae[kt] = b;
  }

  _Float16* scr = scrA + wave * 640;   // 16 rows x 40 halves

  v8f accO[8];
#pragma unroll
  for (int nt = 0; nt < 8; ++nt)
#pragma unroll
    for (int r = 0; r < 8; ++r) accO[nt][r] = 0.0f;

  // ---- MLP-one on s (hidden cols 0..255 of the concat) ----
#pragma unroll
  for (int c = 0; c < 8; ++c)
    mlp_chunk(As[0], As[1], As[2], As[3],
              ldsH + LDS_W1A_H, ldsB, ldsH + LDS_W2_H,
              scr, c, c, col, hi, accO);
  // ---- MLP-two on ef (hidden cols 256..511 of the concat) ----
#pragma unroll
  for (int c = 0; c < 8; ++c)
    mlp_chunk(Ae[0], Ae[1], Ae[2], Ae[3],
              ldsH + LDS_W1B_H, ldsB + 256, ldsH + LDS_W2_H,
              scr, c, c + 8, col, hi, accO);

  // ---- bias, InstanceNorm over 128 channels, ReLU, residual ----
  const float* b2 = ldsB + 512;
#pragma unroll
  for (int nt = 0; nt < 8; ++nt) {
    float bv = b2[nt * 16 + col];
#pragma unroll
    for (int r = 0; r < 8; ++r) accO[nt][r] += bv;
  }

  float sum[8], sq[8];
#pragma unroll
  for (int r = 0; r < 8; ++r) { sum[r] = 0.0f; sq[r] = 0.0f; }
#pragma unroll
  for (int nt = 0; nt < 8; ++nt)
#pragma unroll
    for (int r = 0; r < 8; ++r) {
      float v = accO[nt][r];
      sum[r] += v;
      sq[r]  += v * v;
    }
#pragma unroll
  for (int r = 0; r < 8; ++r)
#pragma unroll
    for (int m = 1; m < 16; m <<= 1) {       // stays inside each 16-lane half
      sum[r] += __shfl_xor(sum[r], m, 32);
      sq[r]  += __shfl_xor(sq[r],  m, 32);
    }

  float mu[8], rs[8];
#pragma unroll
  for (int r = 0; r < 8; ++r) {
    mu[r] = sum[r] * (1.0f / 128.0f);
    float var = sq[r] * (1.0f / 128.0f) - mu[r] * mu[r];
    rs[r] = rsqrtf(var + EPS_IN);
  }

#pragma unroll
  for (int nt = 0; nt < 8; ++nt)
#pragma unroll
    for (int r = 0; r < 8; ++r) {
      int M  = r + hi * 8;
      int eo = eBase + M;
      if (eo < nE) {
        size_t off = (size_t)eo * C_IN + nt * 16 + col;
        float nrm = (accO[nt][r] - mu[r]) * rs[r];
        out[off] = ef[off] + fmaxf(nrm, 0.0f);
      }
    }
}

extern "C" void kernel_launch(void* const* d_in, const int* in_sizes, int n_in,
                              void* d_out, int out_size, void* d_ws, size_t ws_size,
                              hipStream_t stream) {
  const float*     nf  = (const float*)d_in[0];
  const float*     ef  = (const float*)d_in[1];
  const long long* ei  = (const long long*)d_in[2];
  const float*     W1a = (const float*)d_in[3];
  const float*     b1a = (const float*)d_in[4];
  const float*     W2a = (const float*)d_in[5];
  const float*     b2a = (const float*)d_in[6];
  const float*     W1b = (const float*)d_in[7];
  const float*     b1b = (const float*)d_in[8];
  const float*     W2b = (const float*)d_in[9];
  const float*     b2b = (const float*)d_in[10];

  int nE = in_sizes[2] / 2;               // edge_index is (2, E)

  _Float16* wsH = (_Float16*)d_ws;
  float*    wsB = (float*)((char*)d_ws + WS_BIAS_BYTES);

  gnn_prep_kernel<<<128, 256, 0, stream>>>(W1a, b1a, W2a, b2a, W1b, b1b, W2b, b2b,
                                           wsH, wsB);

  hipFuncSetAttribute((const void*)gnn_edge_kernel,
                      hipFuncAttributeMaxDynamicSharedMemorySize, LDS_TOTAL_B);

  int blocks = (nE + EDGES_PER_BLOCK - 1) / EDGES_PER_BLOCK;
  gnn_edge_kernel<<<blocks, THREADS, LDS_TOTAL_B, stream>>>(nf, ef, ei, wsH, wsB,
                                                            (float*)d_out, nE);
}
